// DGCNN_dense_68702296867394
// MI455X (gfx1250) — matrix operand order repaired and checked by reading, not compiled, not run on hardware
//
#include <hip/hip_runtime.h>
#include <hip/hip_bf16.h>
#include <stdint.h>

#define K_NB   20
#define NPTS   2048
#define NBATCH 16
#define NCOLS_E (NBATCH*NPTS*K_NB)   // 655360
#define NCOLS_P (NBATCH*NPTS)        // 32768
#define STAT_NB 512

typedef __attribute__((ext_vector_type(16))) __bf16 bf16x16;
typedef __attribute__((ext_vector_type(8)))  float  f32x8;
typedef __attribute__((ext_vector_type(4)))  unsigned u32x4;
typedef __attribute__((ext_vector_type(8)))  int      i32x8;
typedef __attribute__((ext_vector_type(4)))  int      i32x4;

__device__ __forceinline__ unsigned short f2bf(float f){
    unsigned u = __float_as_uint(f);
    unsigned r = (u + 0x7FFFu + ((u >> 16) & 1u)) >> 16;
    return (unsigned short)r;
}
__device__ __forceinline__ unsigned pack2(float a, float b){
    return (unsigned)f2bf(a) | ((unsigned)f2bf(b) << 16);
}
__device__ __forceinline__ float lrelu(float x){ return x > 0.f ? x : 0.2f * x; }

// ---------------------------------------------------------------------------
// TDM: DMA one B-tile (128 cols x 64 bf16, row stride Cin elements) from
// global into LDS at lds_byte, inserting 16B of padding after every 128B row
// (pad_interval=32 DWORDs -> code 4, pad_amount=4 DWORDs -> code 3) so the
// LDS image lands directly in the 36-uint-pitch layout the WMMA fragment
// loader expects. Descriptor per CDNA5 ISA section 8.3/8.4.
// This toolchain exposes the 6-arg builtin: (g0, g1, g2, g3, g4, cpol).
// ---------------------------------------------------------------------------
__device__ __forceinline__ void tdm_load_tile(const unsigned short* src,
                                              unsigned lds_byte, int Cin)
{
    unsigned long long ga = (unsigned long long)(size_t)src;
    u32x4 g0;
    g0[0] = 1u;                                               // count=1 (valid D#)
    g0[1] = lds_byte;                                         // lds_addr
    g0[2] = (unsigned)ga;                                     // global_addr[31:0]
    g0[3] = (unsigned)((ga >> 32) & 0x01FFFFFFu) | (2u << 30);// ga[56:32] | type=2
    i32x8 g1;
    g1[0] = (int)((1u << 16) | (1u << 20) | (4u << 22) | (3u << 25));
            // data_size=2B, pad_enable, pad_interval=32dw, pad_amount=4dw, mask=0
    g1[1] = (int)(64u << 16);   // tensor_dim0 = 64       (bits 79:48)
    g1[2] = (int)(128u << 16);  // tensor_dim1 = 128      (bits 111:80)
    g1[3] = (int)(64u << 16);   // tile_dim0   = 64       (bits 127:112)
    g1[4] = 128;                // tile_dim1 = 128, tile_dim2 = 0
    g1[5] = Cin;                // tensor_dim0_stride low 32 (elements)
    g1[6] = 0;
    g1[7] = 0;
    i32x4 gz4 = {0, 0, 0, 0};
    i32x8 gz8 = {0, 0, 0, 0, 0, 0, 0, 0};
    __builtin_amdgcn_tensor_load_to_lds(g0, g1, gz4, gz4, gz8, 0);
}

// ---------------------------------------------------------------------------
// WMMA compute core: 64(M) x 16(N per wave) tile, one K-chunk of 64.
// Preload both B fragments and all 8 A fragments, then 8 back-to-back WMMAs
// on 4 independent accumulator chains (lets ds_loads batch under one wait).
// A frag (16x32): lane-half kh: u[j]   -> K=kh*16+2j (16 contiguous K values)
// B frag (32x16): lane-half kh: u[j]   -> K=kh*16+2j (16 contiguous K values)
// ---------------------------------------------------------------------------
__device__ __forceinline__ void wmma_chunk64(const unsigned* sA, const unsigned* sB,
                                             int lane, int wv, f32x8 (&acc)[4])
{
    const int kh  = lane >> 4;
    const int l15 = lane & 15;
    union F { bf16x16 v; unsigned u[8]; };
    F b0, b1, a[8];
    const unsigned* bp = sB + (wv * 16 + l15) * 36 + kh * 8;
#pragma unroll
    for (int j = 0; j < 8; ++j) { b0.u[j] = bp[j]; b1.u[j] = bp[16 + j]; }
#pragma unroll
    for (int m = 0; m < 4; ++m) {
        const unsigned* ap = sA + (m * 16 + l15) * 36 + kh * 8;
#pragma unroll
        for (int j = 0; j < 4; ++j) {
            a[m].u[j]     = ap[j];      a[m].u[4 + j]     = ap[4 + j];
            a[m + 4].u[j] = ap[16 + j]; a[m + 4].u[4 + j] = ap[20 + j];
        }
    }
#pragma unroll
    for (int m = 0; m < 4; ++m)
        acc[m] = __builtin_amdgcn_wmma_f32_16x16x32_bf16(
            false, a[m].v, false, b0.v, (short)0, acc[m], false, false);
#pragma unroll
    for (int m = 0; m < 4; ++m)
        acc[m] = __builtin_amdgcn_wmma_f32_16x16x32_bf16(
            false, a[m + 4].v, false, b1.v, (short)0, acc[m], false, false);
}

// ---------------------------------------------------------------------------
// Generic GEMM: Y[col][Cout] = sum_c W[o][c] * X[col][c], X bf16 [col][Cin].
// B tiles arrive via TDM (double-buffered, overlapped with compute);
// A tile (weights, f32->bf16) is converted by the VALU path.
// grid.x = ncols/128, grid.y = Cout/64. Cin multiple of 64.
// ---------------------------------------------------------------------------
__global__ __launch_bounds__(256)
void gemm_bf16(const unsigned short* __restrict__ X, const float* __restrict__ Wt,
               float* __restrict__ Y, int ncols, int Cin, int Cout)
{
    __shared__ unsigned sA[64 * 36];
    __shared__ unsigned sB[2][128 * 36];
    const int t = threadIdx.x, lane = t & 31, wv = t >> 5;
    const int colbase = blockIdx.x * 128;
    const int mbase   = blockIdx.y * 64;
    const unsigned ldsB0 = (unsigned)(size_t)(void*)&sB[0][0];
    const unsigned ldsB1 = (unsigned)(size_t)(void*)&sB[1][0];
    const unsigned short* Xrow = X + (size_t)colbase * Cin;

    f32x8 acc[4];
#pragma unroll
    for (int m = 0; m < 4; ++m)
#pragma unroll
        for (int r = 0; r < 8; ++r) acc[m][r] = 0.f;

    const int nchunks = Cin >> 6;
    if (wv == 0) tdm_load_tile(Xrow, ldsB0, Cin);     // prologue DMA chunk 0

    int buf = 0;
    for (int kc = 0; kc < nchunks; ++kc) {
        {   // A: 64 rows x 64 k (f32 -> bf16) by VALU
            int row = t >> 2, kb = (t & 3) * 16;
            const float* wp = Wt + (size_t)(mbase + row) * Cin + kc * 64 + kb;
            unsigned* dst = sA + row * 36 + (kb >> 1);
#pragma unroll
            for (int i = 0; i < 16; i += 2) dst[i >> 1] = pack2(wp[i], wp[i + 1]);
        }
        if (wv == 0) {
            if (kc + 1 < nchunks) {                   // prefetch next chunk's B tile
                tdm_load_tile(Xrow + (kc + 1) * 64, buf ? ldsB0 : ldsB1, Cin);
                __builtin_amdgcn_s_wait_tensorcnt(1); // current buffer landed
            } else {
                __builtin_amdgcn_s_wait_tensorcnt(0);
            }
        }
        __syncthreads();
        wmma_chunk64(sA, &sB[buf][0], lane, wv, acc);
        __syncthreads();
        buf ^= 1;
    }

    const int col = colbase + wv * 16 + (lane & 15);
    const int rh  = (lane >> 4) * 8;
#pragma unroll
    for (int m = 0; m < 4; ++m) {
        float* yp = Y + (size_t)col * Cout + mbase + m * 16 + rh;
#pragma unroll
        for (int r = 0; r < 8; ++r) yp[r] = acc[m][r];
    }
}

// ---------------------------------------------------------------------------
// Edge-conv GEMM with in-LDS graph-feature gather (indexed gather + f32->bf16
// conversion, so the B tile is built by the VALU path here).
// xs: (B, CPT, N) point features; nbr: (B,N,K) neighbor idx; W: (64, 2*CPT).
// Feature channel c: c<CPT -> xs[nb]-xs[ctr]; c<2CPT -> xs[ctr]; else 0 pad.
// ---------------------------------------------------------------------------
template<int CPT>
__device__ __forceinline__ float edge_fetch(const float* xs, int bc, int bn, int c)
{
    if (c < CPT)      return xs[bn + c * NPTS] - xs[bc + c * NPTS];
    if (c < 2 * CPT)  return xs[bc + (c - CPT) * NPTS];
    return 0.f;
}

template<int CPT>
__global__ __launch_bounds__(256)
void gemm_edge(const float* __restrict__ xs, const int* __restrict__ nbr,
               const float* __restrict__ Wt, float* __restrict__ Y)
{
    constexpr int CIN_ACT = 2 * CPT;           // 6 or 128
    constexpr int NCHUNK  = (CPT == 3) ? 1 : 2;
    __shared__ unsigned sA[64 * 36];
    __shared__ unsigned sB[128 * 36];
    __shared__ int sCtr[128];
    __shared__ int sNbr[128];
    const int t = threadIdx.x, lane = t & 31, wv = t >> 5;
    const int colbase = blockIdx.x * 128;

    if (t < 128) {
        int col = colbase + t;
        int b   = col / (NPTS * K_NB);
        int rem = col - b * (NPTS * K_NB);
        int n   = rem / K_NB;
        int base = b * CPT * NPTS;
        sCtr[t] = base + n;
        sNbr[t] = base + nbr[col];
    }
    __syncthreads();

    f32x8 acc[4];
#pragma unroll
    for (int m = 0; m < 4; ++m)
#pragma unroll
        for (int r = 0; r < 8; ++r) acc[m][r] = 0.f;

    for (int kc = 0; kc < NCHUNK; ++kc) {
        {   // A: 64 x 64, zero-padded beyond CIN_ACT
            int row = t >> 2, kb = (t & 3) * 16;
            const float* wp = Wt + row * CIN_ACT;
            unsigned* dst = sA + row * 36 + (kb >> 1);
#pragma unroll
            for (int i = 0; i < 16; i += 2) {
                int k0 = kc * 64 + kb + i;
                float v0 = (k0     < CIN_ACT) ? wp[k0]     : 0.f;
                float v1 = (k0 + 1 < CIN_ACT) ? wp[k0 + 1] : 0.f;
                dst[i >> 1] = pack2(v0, v1);
            }
        }
        {   // B: gather 128 cols x 64 channels
            int cl = t >> 1;
            int c0 = kc * 64 + (t & 1) * 32;
            int bc = sCtr[cl], bn = sNbr[cl];
            unsigned* dst = sB + cl * 36 + (t & 1) * 16;
#pragma unroll
            for (int i = 0; i < 32; i += 2) {
                float v0 = edge_fetch<CPT>(xs, bc, bn, c0 + i);
                float v1 = edge_fetch<CPT>(xs, bc, bn, c0 + i + 1);
                dst[i >> 1] = pack2(v0, v1);
            }
        }
        __syncthreads();
        wmma_chunk64(sA, sB, lane, wv, acc);
        __syncthreads();
    }
    const int col = colbase + wv * 16 + (lane & 15);
    const int rh  = (lane >> 4) * 8;
#pragma unroll
    for (int m = 0; m < 4; ++m) {
        float* yp = Y + (size_t)col * 64 + m * 16 + rh;
#pragma unroll
        for (int r = 0; r < 8; ++r) yp[r] = acc[m][r];
    }
}

// ---------------------------------------------------------------------------
// Top-k (k=20) nearest neighbors. 1 wave per point, pw row in LDS,
// 20 iterative wave-argmax extractions (stable: ties -> lower index).
// ---------------------------------------------------------------------------
template<int CPT>
__global__ __launch_bounds__(256)
void topk_kernel(const float* __restrict__ xs, int* __restrict__ nbr)
{
    __shared__ float spw[8][NPTS];
    __shared__ float sctr[8][64];
    const int t = threadIdx.x, lane = t & 31, wv = t >> 5;
    const int p = blockIdx.x * 8 + wv;
    const int b = p / NPTS, n = p % NPTS;
    const float* xb = xs + (size_t)b * CPT * NPTS;

    for (int c = lane; c < CPT; c += 32) sctr[wv][c] = xb[c * NPTS + n];
    __syncthreads();

    float xxn = 0.f;
    for (int c = 0; c < CPT; ++c) { float v = sctr[wv][c]; xxn += v * v; }

    for (int m0 = 0; m0 < NPTS; m0 += 32) {
        int m = m0 + lane;
        float dot = 0.f, s2 = 0.f;
        for (int c = 0; c < CPT; ++c) {
            float v = xb[c * NPTS + m];
            dot += v * sctr[wv][c]; s2 += v * v;
        }
        spw[wv][m] = 2.f * dot - s2 - xxn;   // -||xi-xj||^2
    }
    __syncthreads();

    for (int tt = 0; tt < K_NB; ++tt) {
        float best = -3.4e38f; int bi = NPTS;
        for (int i = lane; i < NPTS; i += 32) {
            float v = spw[wv][i];
            if (v > best || (v == best && i < bi)) { best = v; bi = i; }
        }
#pragma unroll
        for (int off = 16; off; off >>= 1) {
            float ov = __shfl_xor(best, off, 32);
            int   oi = __shfl_xor(bi,   off, 32);
            if (ov > best || (ov == best && oi < bi)) { best = ov; bi = oi; }
        }
        if (lane == 0) {
            nbr[(size_t)p * K_NB + tt] = bi;
            spw[wv][bi] = -3.4e38f;
        }
        __syncthreads();
    }
}

// ---------------------------------------------------------------------------
// Deterministic per-channel stats (sum, sumsq) over columns. Two stages.
// C in {64,256,512,1024} (powers of two).
// ---------------------------------------------------------------------------
__global__ __launch_bounds__(256)
void stats_partial(const float* __restrict__ Y, int ncols, int C, float* __restrict__ part)
{
    __shared__ float ls[256], lq[256];
    const int t = threadIdx.x;
    const int P  = (C < 256) ? C : 256;
    const int R  = 256 / P;
    const int CH = (C + 255) / 256;
    const int tc = t % P, tr = t / P;
    const int cpb = (ncols + gridDim.x - 1) / gridDim.x;
    const int c0 = blockIdx.x * cpb;
    const int c1 = min(ncols, c0 + cpb);
    float s[4] = {0,0,0,0}, q[4] = {0,0,0,0};
    for (int col = c0 + tr; col < c1; col += R) {
        const float* yp = Y + (size_t)col * C + tc;
        for (int u = 0; u < CH; ++u) { float v = yp[u * 256]; s[u] += v; q[u] += v * v; }
    }
    for (int u = 0; u < CH; ++u) {
        ls[t] = s[u]; lq[t] = q[u];
        __syncthreads();
        if (tr == 0) {
            float ss = s[u], qq = q[u];
            for (int r = 1; r < R; ++r) { ss += ls[tc + r * P]; qq += lq[tc + r * P]; }
            int ch = tc + u * 256;
            part[(size_t)blockIdx.x * 2 * C + ch]     = ss;
            part[(size_t)blockIdx.x * 2 * C + C + ch] = qq;
        }
        __syncthreads();
    }
}

__global__ void stats_reduce(const float* __restrict__ part, int C, int nb,
                             float* __restrict__ stats)
{
    int c = blockIdx.x * blockDim.x + threadIdx.x;
    if (c >= C) return;
    float s = 0.f, q = 0.f;
    for (int b = 0; b < nb; ++b) {
        s += part[(size_t)b * 2 * C + c];
        q += part[(size_t)b * 2 * C + C + c];
    }
    stats[c] = s; stats[C + c] = q;
}

// ---------------------------------------------------------------------------
// BN(+LeakyReLU) apply variants
// ---------------------------------------------------------------------------
__global__ __launch_bounds__(256)
void bn_act_bf16(const float* __restrict__ Y, const float* __restrict__ stats,
                 const float* __restrict__ g, const float* __restrict__ bet,
                 unsigned short* __restrict__ out, int ncols, int C)
{
    __shared__ float ssc[1024], ssh[1024];
    const int t = threadIdx.x;
    const float invM = 1.f / (float)ncols;
    for (int c = t; c < C; c += 256) {
        float mean = stats[c] * invM;
        float var  = stats[C + c] * invM - mean * mean;
        float sc   = g[c] * rsqrtf(var + 1e-5f);
        ssc[c] = sc; ssh[c] = bet[c] - mean * sc;
    }
    __syncthreads();
    size_t total  = (size_t)ncols * C;
    size_t stride = (size_t)gridDim.x * 256;
    for (size_t e = (size_t)blockIdx.x * 256 + t; e < total; e += stride) {
        int c = (int)(e & (size_t)(C - 1));
        out[e] = f2bf(lrelu(Y[e] * ssc[c] + ssh[c]));
    }
}

__global__ __launch_bounds__(256)
void bn_maxk(const float* __restrict__ Y, const float* __restrict__ stats,
             const float* __restrict__ g, const float* __restrict__ bet,
             float* __restrict__ xout, int ncols)
{
    __shared__ float ssc[64], ssh[64];
    const int t = threadIdx.x;
    const float invM = 1.f / (float)ncols;
    if (t < 64) {
        float mean = stats[t] * invM;
        float var  = stats[64 + t] * invM - mean * mean;
        float sc   = g[t] * rsqrtf(var + 1e-5f);
        ssc[t] = sc; ssh[t] = bet[t] - mean * sc;
    }
    __syncthreads();
    const int p  = blockIdx.x * 4 + (t >> 6);
    const int ch = t & 63;
    const int b = p / NPTS, n = p % NPTS;
    const float* yp = Y + (size_t)p * K_NB * 64 + ch;
    float m = -3.4e38f;
#pragma unroll
    for (int k = 0; k < K_NB; ++k)
        m = fmaxf(m, lrelu(yp[k * 64] * ssc[ch] + ssh[ch]));
    xout[((size_t)b * 64 + ch) * NPTS + n] = m;
}

__global__ __launch_bounds__(256)
void bn_maxn(const float* __restrict__ Y, const float* __restrict__ stats,
             const float* __restrict__ g, const float* __restrict__ bet,
             float* __restrict__ gmax)
{
    const int ch = blockIdx.y * 256 + threadIdx.x;  // C = 1024
    const int b  = blockIdx.x;
    const float invM = 1.f / (float)NCOLS_P;
    float mean = stats[ch] * invM;
    float var  = stats[1024 + ch] * invM - mean * mean;
    float sc   = g[ch] * rsqrtf(var + 1e-5f);
    float sh   = bet[ch] - mean * sc;
    const float* yp = Y + (size_t)b * NPTS * 1024 + ch;
    float m = -3.4e38f;
    for (int n = 0; n < NPTS; ++n)
        m = fmaxf(m, lrelu(yp[(size_t)n * 1024] * sc + sh));
    gmax[b * 1024 + ch] = m;
}

__global__ __launch_bounds__(256)
void bn_out_c(const float* __restrict__ Y, const float* __restrict__ stats,
              const float* __restrict__ g, const float* __restrict__ bet,
              float* __restrict__ out)
{
    __shared__ float ssc[256], ssh[256];
    const int t = threadIdx.x;
    const float invM = 1.f / (float)NCOLS_P;
    {
        float mean = stats[t] * invM;
        float var  = stats[256 + t] * invM - mean * mean;
        float sc   = g[t] * rsqrtf(var + 1e-5f);
        ssc[t] = sc; ssh[t] = bet[t] - mean * sc;
    }
    __syncthreads();
    size_t total  = (size_t)NCOLS_P * 256;
    size_t stride = (size_t)gridDim.x * 256;
    for (size_t e = (size_t)blockIdx.x * 256 + t; e < total; e += stride) {
        int ch = (int)(e & 255);
        size_t col = e >> 8;
        int b = (int)(col >> 11), n = (int)(col & 2047);
        out[((size_t)b * 256 + ch) * NPTS + n] = Y[e] * ssc[ch] + ssh[ch];
    }
}

// ---------------------------------------------------------------------------
// Packing / concatenation helpers
// ---------------------------------------------------------------------------
__global__ __launch_bounds__(256)
void pack_x123(const float* __restrict__ x1, const float* __restrict__ x2,
               const float* __restrict__ x3, unsigned short* __restrict__ out)
{
    size_t e = (size_t)blockIdx.x * 256 + threadIdx.x;   // exact: 32768*192
    int col = (int)(e / 192), ch = (int)(e % 192);
    int b = col >> 11, n = col & 2047;
    const float* src = (ch < 64) ? x1 : ((ch < 128) ? x2 : x3);
    out[e] = f2bf(src[((size_t)b * 64 + (ch & 63)) * NPTS + n]);
}

__global__ __launch_bounds__(256)
void build_xz(const float* __restrict__ gmax, const float* __restrict__ x1,
              const float* __restrict__ x2, const float* __restrict__ x3,
              float* __restrict__ outf, unsigned short* __restrict__ out16)
{
    size_t e = (size_t)blockIdx.x * 256 + threadIdx.x;   // exact: 16*1216*2048
    int b   = (int)(e / (1216 * 2048));
    int rem = (int)(e - (size_t)b * (1216 * 2048));
    int ch = rem >> 11, n = rem & 2047;
    float v;
    if (ch < 1024)      v = gmax[b * 1024 + ch];
    else if (ch < 1088) v = x1[((size_t)b * 64 + (ch - 1024)) * NPTS + n];
    else if (ch < 1152) v = x2[((size_t)b * 64 + (ch - 1088)) * NPTS + n];
    else                v = x3[((size_t)b * 64 + (ch - 1152)) * NPTS + n];
    outf[e] = v;
    out16[((size_t)(b * NPTS + n)) * 1216 + ch] = f2bf(v);
}

// ---------------------------------------------------------------------------
// Tiny pointwise MLP layers on the global vector (N==1): per-channel BN over
// the 16 batch values, entirely thread-local -> deterministic and cheap.
// ---------------------------------------------------------------------------
__global__ void pw_layer(const float* __restrict__ in, const float* __restrict__ Wt,
                         const float* __restrict__ g, const float* __restrict__ bet,
                         float* __restrict__ out, int Cin, int Cout, int act)
{
    int o = blockIdx.x * blockDim.x + threadIdx.x;
    if (o >= Cout) return;
    float y[NBATCH];
    const float* wp = Wt + (size_t)o * Cin;
    for (int bb = 0; bb < NBATCH; ++bb) {
        const float* ip = in + (size_t)bb * Cin;
        float s = 0.f;
        for (int c = 0; c < Cin; ++c) s += wp[c] * ip[c];
        y[bb] = s;
    }
    float mean = 0.f;
    for (int bb = 0; bb < NBATCH; ++bb) mean += y[bb];
    mean *= (1.f / NBATCH);
    float var = 0.f;
    for (int bb = 0; bb < NBATCH; ++bb) { float d = y[bb] - mean; var += d * d; }
    var *= (1.f / NBATCH);
    float sc = g[o] * rsqrtf(var + 1e-5f), sh = bet[o] - mean * sc;
    for (int bb = 0; bb < NBATCH; ++bb) {
        float v = y[bb] * sc + sh;
        if (act) v = lrelu(v);
        out[bb * Cout + o] = v;
    }
}

// ---------------------------------------------------------------------------
// Host orchestration
// ---------------------------------------------------------------------------
extern "C" void kernel_launch(void* const* d_in, const int* in_sizes, int n_in,
                              void* d_out, int out_size, void* d_ws, size_t ws_size,
                              hipStream_t stream)
{
    (void)in_sizes; (void)n_in; (void)out_size; (void)ws_size;
    const float* x   = (const float*)d_in[0];
    const float* w1  = (const float*)d_in[1];
    const float* w2  = (const float*)d_in[2];
    const float* w3  = (const float*)d_in[3];
    const float* w4  = (const float*)d_in[4];
    const float* w5  = (const float*)d_in[5];
    const float* w6  = (const float*)d_in[6];
    const float* pw1 = (const float*)d_in[7];
    const float* pw2 = (const float*)d_in[8];
    const float* pw3 = (const float*)d_in[9];
    const float* cw1 = (const float*)d_in[10];
    const float* cw2 = (const float*)d_in[11];
    const float* cw3 = (const float*)d_in[12];
    const float* g1 = (const float*)d_in[13]; const float* b1 = (const float*)d_in[14];
    const float* g2 = (const float*)d_in[15]; const float* b2 = (const float*)d_in[16];
    const float* g3 = (const float*)d_in[17]; const float* b3 = (const float*)d_in[18];
    const float* g4 = (const float*)d_in[19]; const float* b4 = (const float*)d_in[20];
    const float* g5 = (const float*)d_in[21]; const float* b5 = (const float*)d_in[22];
    const float* g6 = (const float*)d_in[23]; const float* b6 = (const float*)d_in[24];
    const float* pg1 = (const float*)d_in[25]; const float* pb1 = (const float*)d_in[26];
    const float* pg2 = (const float*)d_in[27]; const float* pb2 = (const float*)d_in[28];
    const float* pg3 = (const float*)d_in[29]; const float* pb3 = (const float*)d_in[30];
    const float* cg1 = (const float*)d_in[31]; const float* cb1 = (const float*)d_in[32];
    const float* cg2 = (const float*)d_in[33]; const float* cb2 = (const float*)d_in[34];
    const float* cg3 = (const float*)d_in[35]; const float* cb3 = (const float*)d_in[36];

    float* outFZ = (float*)d_out;                              // (16,256)
    float* outC  = outFZ + (size_t)NBATCH * 256;               // (16,256,2048)
    float* outXZ = outC  + (size_t)NBATCH * 256 * NPTS;        // (16,1216,2048)

    // workspace layout
    char* ws = (char*)d_ws;
    size_t off = 0;
    auto alloc = [&](size_t bytes) { size_t r = off; off += (bytes + 255) & ~(size_t)255; return r; };
    int*   idx   = (int*)  (ws + alloc((size_t)NCOLS_E * 4));
    float* yraw  = (float*)(ws + alloc((size_t)NCOLS_E * 64 * 4));       // reused for all GEMM outputs
    unsigned short* hbf = (unsigned short*)(ws + alloc((size_t)NCOLS_E * 64 * 2)); // h / xcat / xz16
    float* x1 = (float*)(ws + alloc((size_t)NBATCH * 64 * NPTS * 4));
    float* x2 = (float*)(ws + alloc((size_t)NBATCH * 64 * NPTS * 4));
    float* x3 = (float*)(ws + alloc((size_t)NBATCH * 64 * NPTS * 4));
    float* gmax = (float*)(ws + alloc((size_t)NBATCH * 1024 * 4));
    float* z1   = (float*)(ws + alloc((size_t)NBATCH * 512 * 4));
    float* z2   = (float*)(ws + alloc((size_t)NBATCH * 256 * 4));
    float* stats = (float*)(ws + alloc(2048 * 4));
    float* part  = (float*)(ws + alloc((size_t)STAT_NB * 2048 * 4));
    unsigned short* c1bf = (unsigned short*)(ws + alloc((size_t)NCOLS_P * 512 * 2));
    unsigned short* c2bf = (unsigned short*)(ws + alloc((size_t)NCOLS_P * 256 * 2));

    const dim3 blk(256);
    const int gE = NCOLS_E / 128;   // 5120 col-tiles for edge convs
    const int gP = NCOLS_P / 128;   // 256  col-tiles for point convs

    auto do_stats = [&](int ncols, int C) {
        stats_partial<<<STAT_NB, blk, 0, stream>>>(yraw, ncols, C, part);
        stats_reduce<<<(C + 255) / 256, dim3(C < 256 ? C : 256), 0, stream>>>(part, C, STAT_NB, stats);
    };

    // ---- Stage 1 -----------------------------------------------------------
    topk_kernel<3><<<NBATCH * NPTS / 8, blk, 0, stream>>>(x, idx);
    gemm_edge<3><<<gE, blk, 0, stream>>>(x, idx, w1, yraw);
    do_stats(NCOLS_E, 64);
    bn_act_bf16<<<4096, blk, 0, stream>>>(yraw, stats, g1, b1, hbf, NCOLS_E, 64);
    gemm_bf16<<<dim3(gE, 1), blk, 0, stream>>>(hbf, w2, yraw, NCOLS_E, 64, 64);
    do_stats(NCOLS_E, 64);
    bn_maxk<<<NBATCH * NPTS / 4, blk, 0, stream>>>(yraw, stats, g2, b2, x1, NCOLS_E);

    // ---- Stage 2 -----------------------------------------------------------
    topk_kernel<64><<<NBATCH * NPTS / 8, blk, 0, stream>>>(x1, idx);
    gemm_edge<64><<<gE, blk, 0, stream>>>(x1, idx, w3, yraw);
    do_stats(NCOLS_E, 64);
    bn_act_bf16<<<4096, blk, 0, stream>>>(yraw, stats, g3, b3, hbf, NCOLS_E, 64);
    gemm_bf16<<<dim3(gE, 1), blk, 0, stream>>>(hbf, w4, yraw, NCOLS_E, 64, 64);
    do_stats(NCOLS_E, 64);
    bn_maxk<<<NBATCH * NPTS / 4, blk, 0, stream>>>(yraw, stats, g4, b4, x2, NCOLS_E);

    // ---- Stage 3 -----------------------------------------------------------
    topk_kernel<64><<<NBATCH * NPTS / 8, blk, 0, stream>>>(x2, idx);
    gemm_edge<64><<<gE, blk, 0, stream>>>(x2, idx, w5, yraw);
    do_stats(NCOLS_E, 64);
    bn_maxk<<<NBATCH * NPTS / 4, blk, 0, stream>>>(yraw, stats, g5, b5, x3, NCOLS_E);

    // ---- w6 -> x4 -> global max -------------------------------------------
    pack_x123<<<(size_t)NCOLS_P * 192 / 256, blk, 0, stream>>>(x1, x2, x3, hbf);
    gemm_bf16<<<dim3(gP, 16), blk, 0, stream>>>(hbf, w6, yraw, NCOLS_P, 192, 1024);
    do_stats(NCOLS_P, 1024);
    bn_maxn<<<dim3(NBATCH, 4), blk, 0, stream>>>(yraw, stats, g6, b6, gmax);

    // ---- pw MLP -> feature_z ----------------------------------------------
    pw_layer<<<2, blk, 0, stream>>>(gmax, pw1, pg1, pb1, z1, 1024, 512, 1);
    pw_layer<<<1, blk, 0, stream>>>(z1, pw2, pg2, pb2, z2, 512, 256, 1);
    pw_layer<<<1, blk, 0, stream>>>(z2, pw3, pg3, pb3, outFZ, 256, 256, 0);

    // ---- x_z (output 3) + bf16 copy for cw1 --------------------------------
    build_xz<<<(size_t)NBATCH * 1216 * NPTS / 256, blk, 0, stream>>>(gmax, x1, x2, x3, outXZ, hbf);

    // ---- cw chain -> c (output 2) -----------------------------------------
    gemm_bf16<<<dim3(gP, 8), blk, 0, stream>>>(hbf, cw1, yraw, NCOLS_P, 1216, 512);
    do_stats(NCOLS_P, 512);
    bn_act_bf16<<<4096, blk, 0, stream>>>(yraw, stats, cg1, cb1, c1bf, NCOLS_P, 512);
    gemm_bf16<<<dim3(gP, 4), blk, 0, stream>>>(c1bf, cw2, yraw, NCOLS_P, 512, 256);
    do_stats(NCOLS_P, 256);
    bn_act_bf16<<<4096, blk, 0, stream>>>(yraw, stats, cg2, cb2, c2bf, NCOLS_P, 256);
    gemm_bf16<<<dim3(gP, 4), blk, 0, stream>>>(c2bf, cw3, yraw, NCOLS_P, 256, 256);
    do_stats(NCOLS_P, 256);
    bn_out_c<<<8192, blk, 0, stream>>>(yraw, stats, cg3, cb3, outC);
}
